// EncoderLayer_75290776699351
// MI455X (gfx1250) — matrix operand order repaired and checked by reading, not compiled, not run on hardware
//
#include <hip/hip_runtime.h>
#include <hip/hip_bf16.h>

// ---------------------------------------------------------------------------
// Transformer encoder layer for MI455X (gfx1250), bf16 WMMA pipeline.
// B=2, S=2048, D=768, H=3072, 12 heads x 64.
// - All GEMMs + attention on v_wmma_f32_16x16x32_bf16 (f32 accumulate).
// - GEMM A tile: GLOBAL_LOAD_ASYNC_TO_LDS_B128 (ASYNCcnt engine).
// - GEMM B tile: TENSOR_LOAD_TO_LDS (TDM, 6-arg builtin) with hardware LDS
//   padding reproducing the 80B padded row stride; s_wait_tensorcnt sync.
// - Both double-buffered: stage k+1 copies overlap WMMA on stage k; pipeline
//   manually unrolled x2 so buffer indices are compile-time constants.
// - Flash attention: online softmax, async-staged K/V, no score matrix.
// ---------------------------------------------------------------------------

#define DMODEL   768
#define HIDDEN   3072
#define NHEADS   12
#define HEADDIM  64
#define SEQ      2048
#define BATCH    2
#define ROWS     (BATCH * SEQ)        // 4096
#define D3       (3 * DMODEL)         // 2304

typedef __attribute__((ext_vector_type(16))) __bf16        v16bf;
typedef __attribute__((ext_vector_type(8)))  float         v8f;
typedef __attribute__((ext_vector_type(4)))  unsigned int  v4u;
typedef __attribute__((ext_vector_type(4)))  int           v4i;
typedef __attribute__((ext_vector_type(8)))  int           v8i;

typedef __attribute__((address_space(1))) v4i v4i_g;   // global int4
typedef __attribute__((address_space(3))) v4i v4i_s;   // LDS int4

union FragU {                 // one WMMA A/B operand: 16 bf16 = 32 bytes
    v16bf v;
    v4u   q[2];
    unsigned short s[16];
};

__device__ __forceinline__ unsigned short f32_to_bf16(float f) {
    unsigned int u = __float_as_uint(f);
    unsigned int r = u + 0x7FFFu + ((u >> 16) & 1u);   // round-to-nearest-even
    return (unsigned short)(r >> 16);
}

__device__ __forceinline__ v8f wmma_bf16(const FragU& a, const FragU& b, v8f c) {
#if defined(__HIP_DEVICE_COMPILE__)
    return __builtin_amdgcn_wmma_f32_16x16x32_bf16(false, a.v, false, b.v,
                                                   (short)0, c, false, false);
#else
    return c;
#endif
}

// ---- async global->LDS staging (ASYNCcnt engine), with safe fallback ----
#if defined(__HIP_DEVICE_COMPILE__) && __has_builtin(__builtin_amdgcn_global_load_async_to_lds_b128)
#define HAVE_ASYNC_LDS 1
#endif
// ---- TDM tensor DMA (TENSORcnt engine), with safe fallback ----
#if defined(__HIP_DEVICE_COMPILE__) && __has_builtin(__builtin_amdgcn_tensor_load_to_lds)
#define HAVE_TDM 1
#endif

__device__ __forceinline__ void stage_b128(const unsigned short* g, unsigned short* l) {
#if defined(HAVE_ASYNC_LDS)
    __builtin_amdgcn_global_load_async_to_lds_b128(
        (v4i_g*)(unsigned long long)(size_t)g,
        (v4i_s*)(unsigned int)(size_t)l, 0, 0);
#else
    *(v4u*)l = *(const v4u*)g;
#endif
}
__device__ __forceinline__ void stage_wait() {
#if defined(HAVE_ASYNC_LDS)
#if __has_builtin(__builtin_amdgcn_s_wait_asynccnt)
    __builtin_amdgcn_s_wait_asynccnt(0);
#else
    asm volatile("s_wait_asynccnt 0x0" ::: "memory");
#endif
#endif
}
__device__ __forceinline__ void tdm_wait() {
#if defined(HAVE_TDM)
#if __has_builtin(__builtin_amdgcn_s_wait_tensorcnt)
    __builtin_amdgcn_s_wait_tensorcnt(0);
#else
    asm volatile("s_wait_tensorcnt 0x0" ::: "memory");
#endif
#endif
}

// TDM: DMA a 2D tile (nrows x rowlen bf16) from a row-major tensor into LDS,
// with hardware padding of 16B after every 64B -> LDS row stride 80B.
// D# layout per CDNA5 ISA ch.8 (group0: flags/lds/global addr; group1: dims).
__device__ __forceinline__ void tdm_load_2d(const unsigned short* gbase,
                                            const unsigned short* lbase,
                                            int rowlen, int nrows,
                                            int row_stride, int tensor_rows) {
#if defined(HAVE_TDM)
    unsigned long long ga = (unsigned long long)(size_t)gbase;
    v4u g0;
    g0[0] = 1u;                                          // count=1, user D#
    g0[1] = (unsigned int)(size_t)lbase;                 // lds_addr
    g0[2] = (unsigned int)(ga & 0xFFFFFFFFull);          // global_addr[31:0]
    g0[3] = (unsigned int)((ga >> 32) & 0x01FFFFFFull)   // global_addr[56:32]
          | 0x80000000u;                                 // type=2 ("image")
    v8i g1;
    g1[0] = (int)((1u << 16)      // data_size = 2 bytes
                | (1u << 20)      // pad_enable
                | (3u << 22)      // pad_interval: 16 DWORDs (64B)
                | (3u << 25));    // pad_amount:   4 DWORDs (16B)
    g1[1] = (int)(((unsigned)row_stride & 0xFFFFu) << 16);   // tensor_dim0[15:0]
    g1[2] = (int)(((unsigned)tensor_rows & 0xFFFFu) << 16);  // tensor_dim1[15:0]
    g1[3] = (int)(((unsigned)rowlen & 0xFFFFu) << 16);       // tile_dim0
    g1[4] = (int)((unsigned)nrows & 0xFFFFu);                // tile_dim1
    g1[5] = row_stride;                                      // tensor_dim0_stride
    g1[6] = 0;
    g1[7] = 0;
    v4i z4;
    z4[0] = 0; z4[1] = 0; z4[2] = 0; z4[3] = 0;
    v8i z8;
#pragma unroll
    for (int i = 0; i < 8; ++i) z8[i] = 0;
    __builtin_amdgcn_tensor_load_to_lds(g0, g1, z4, z4, z8, 0);
#else
    (void)gbase; (void)lbase; (void)rowlen; (void)nrows;
    (void)row_stride; (void)tensor_rows;
#endif
}

__device__ __forceinline__ float wave_red_sum(float v) {
#pragma unroll
    for (int off = 16; off > 0; off >>= 1) v += __shfl_xor(v, off, 32);
    return v;
}

// ---------------------------------------------------------------------------
// f32 -> bf16 weight repack with TRANSPOSE: src rows x cols (row-major, ld=sld)
// -> dst[c * dld + r] (bf16).  Lets GEMM consume B as N x K row-major.
// ---------------------------------------------------------------------------
__global__ void cvt_t_f32_bf16(const float* __restrict__ src,
                               unsigned short* __restrict__ dst,
                               int rows, int cols, int sld, int dld) {
    int idx = blockIdx.x * 256 + threadIdx.x;
    if (idx >= rows * cols) return;
    int r = idx / cols, c = idx - r * cols;
    dst[(size_t)c * dld + r] = f32_to_bf16(src[(size_t)r * sld + c]);
}

// ---------------------------------------------------------------------------
// V transpose: qkv (ROWS x 3D, bf16) -> vT[((b*NH+h)*64 + d) * SEQ + s]
// ---------------------------------------------------------------------------
__global__ void transpose_v(const unsigned short* __restrict__ qkv,
                            unsigned short* __restrict__ vT) {
    int idx = blockIdx.x * 256 + threadIdx.x;
    if (idx >= ROWS * DMODEL) return;
    int row = idx / DMODEL, col = idx - row * DMODEL;
    int b = row / SEQ, s = row - b * SEQ;
    int h = col / HEADDIM, d = col - h * HEADDIM;
    vT[(((size_t)b * NHEADS + h) * HEADDIM + d) * SEQ + s] =
        qkv[(size_t)row * D3 + 2 * DMODEL + col];
}

// ---------------------------------------------------------------------------
// LayerNorm (torch semantics: unbiased std, (std + eps)); D = 768 = 256*3.
// ---------------------------------------------------------------------------
__global__ void __launch_bounds__(256)
layernorm_bf16(const float* __restrict__ x, const float* __restrict__ gamma,
               const float* __restrict__ beta, unsigned short* __restrict__ out) {
    const int row  = blockIdx.x;
    const int tid  = threadIdx.x;
    const int lane = tid & 31, wid = tid >> 5;
    const float* xr = x + (size_t)row * DMODEL;

    float v[3];
    float s = 0.f;
#pragma unroll
    for (int i = 0; i < 3; ++i) { v[i] = xr[tid + i * 256]; s += v[i]; }

    __shared__ float red[8];
    s = wave_red_sum(s);
    if (lane == 0) red[wid] = s;
    __syncthreads();
    float tot = 0.f;
#pragma unroll
    for (int w = 0; w < 8; ++w) tot += red[w];
    const float mean = tot * (1.0f / DMODEL);

    float d2 = 0.f;
#pragma unroll
    for (int i = 0; i < 3; ++i) { float d = v[i] - mean; d2 += d * d; }
    __syncthreads();
    d2 = wave_red_sum(d2);
    if (lane == 0) red[wid] = d2;
    __syncthreads();
    float tot2 = 0.f;
#pragma unroll
    for (int w = 0; w < 8; ++w) tot2 += red[w];
    const float inv = 1.0f / (sqrtf(tot2 * (1.0f / (DMODEL - 1))) + 1e-7f);

    unsigned short* orow = out + (size_t)row * DMODEL;
#pragma unroll
    for (int i = 0; i < 3; ++i) {
        int c = tid + i * 256;
        orow[c] = f32_to_bf16(gamma[c] * ((v[i] - mean) * inv) + beta[c]);
    }
}

// ---------------------------------------------------------------------------
// bf16 WMMA GEMM:  out = A(MxK rowmajor) * Wt(NxK rowmajor)^T
// Block tile 128x256, BK=32. 8 waves (2x4), wave tile 64x64 = 4x4 WMMA tiles.
// A tile -> async engine; B tile -> TDM (wave 0 issues, hardware pads rows).
// Double-buffered, pipeline unrolled x2 (K % 64 == 0 for all layers).
// ---------------------------------------------------------------------------
#define GBM 128
#define GBN 256
#define GBK 32
#define GLD 40     // ushort stride = 80B = 64B data + 16B TDM pad

template <bool RELU, bool HAS_RES, bool OUT_BF>
__global__ void __launch_bounds__(256)
gemm_bf16_wmma(const unsigned short* __restrict__ A,
               const unsigned short* __restrict__ Wt,
               const float* __restrict__ res,
               float* __restrict__ outF,
               unsigned short* __restrict__ outB,
               int M, int N, int K) {
    __shared__ __attribute__((aligned(16))) unsigned short As[2][GBM * GLD];
    __shared__ __attribute__((aligned(16))) unsigned short Bs[2][GBN * GLD];

    const int tid   = threadIdx.x;
    const int lane  = tid & 31, wid = tid >> 5;
    const int wm    = wid >> 2, wn = wid & 3;       // 2x4 wave grid
    const int lmod  = lane & 15, lhalf = lane >> 4;
    const int m0    = blockIdx.y * GBM;
    const int n0    = blockIdx.x * GBN;

    const int a_row = tid >> 1, a_colb = (tid & 1) * 16;   // 2 x b128 each

    auto stage_tiles = [&](int k0, int buf) {
        // A tile 128x32: per-lane async b128 copies
        const unsigned short* ag = A + (size_t)(m0 + a_row) * K + k0 + a_colb;
#pragma unroll
        for (int i = 0; i < 2; ++i)
            stage_b128(ag + i * 8, &As[buf][a_row * GLD + a_colb + i * 8]);
        if (k0 + GBK < K) __builtin_prefetch(ag + GBK, 0, 1);
        // B tile 256x32: one TDM descriptor per workgroup (wave 0 issues)
#if defined(HAVE_TDM)
        if (wid == 0)
            tdm_load_2d(Wt + (size_t)n0 * K + k0, &Bs[buf][0],
                        GBK, GBN, K, N);
#else
        const unsigned short* bg = Wt + (size_t)(n0 + tid) * K + k0;
#pragma unroll
        for (int i = 0; i < 4; ++i)
            stage_b128(bg + i * 8, &Bs[buf][tid * GLD + i * 8]);
#endif
    };

    v8f acc[4][4];
#pragma unroll
    for (int mi = 0; mi < 4; ++mi)
#pragma unroll
        for (int ni = 0; ni < 4; ++ni)
#pragma unroll
            for (int e = 0; e < 8; ++e) acc[mi][ni][e] = 0.0f;

    auto compute = [&](int buf) {
        FragU af[4], bfr[4];
#pragma unroll
        for (int mi = 0; mi < 4; ++mi) {
            int base = (wm * 64 + mi * 16 + lmod) * GLD + lhalf * 8;
            af[mi].q[0] = *(const v4u*)&As[buf][base];
            af[mi].q[1] = *(const v4u*)&As[buf][base + 16];
        }
#pragma unroll
        for (int ni = 0; ni < 4; ++ni) {
            int base = (wn * 64 + ni * 16 + lmod) * GLD + lhalf * 8;
            bfr[ni].q[0] = *(const v4u*)&Bs[buf][base];
            bfr[ni].q[1] = *(const v4u*)&Bs[buf][base + 16];
        }
#pragma unroll
        for (int mi = 0; mi < 4; ++mi)
#pragma unroll
            for (int ni = 0; ni < 4; ++ni)
                acc[mi][ni] = wmma_bf16(af[mi], bfr[ni], acc[mi][ni]);
    };

    stage_tiles(0, 0);                       // prologue
    for (int k0 = 0; k0 < K; k0 += 2 * GBK) {
        stage_wait(); tdm_wait();
        __syncthreads();
        stage_tiles(k0 + GBK, 1);            // overlap with WMMA on buf 0
        compute(0);

        stage_wait(); tdm_wait();
        __syncthreads();
        if (k0 + 2 * GBK < K) stage_tiles(k0 + 2 * GBK, 0);
        compute(1);
    }

    // ---- epilogue: C layout VGPR r -> M = r + 8*lhalf, N = lmod ----
#pragma unroll
    for (int mi = 0; mi < 4; ++mi) {
#pragma unroll
        for (int ni = 0; ni < 4; ++ni) {
            int rb  = m0 + wm * 64 + mi * 16 + lhalf * 8;
            int col = n0 + wn * 64 + ni * 16 + lmod;
#pragma unroll
            for (int r = 0; r < 8; ++r) {
                float vv = acc[mi][ni][r];
                if (RELU) vv = fmaxf(vv, 0.0f);
                size_t idx = (size_t)(rb + r) * N + col;
                if (HAS_RES) vv += res[idx];
                if (OUT_BF) outB[idx] = f32_to_bf16(vv);
                else        outF[idx] = vv;
            }
        }
    }
}

// ---------------------------------------------------------------------------
// Flash attention: one wave32 per (batch, head, 16-row q tile).
// K staged row-major (= B^T fragment order for Q.K^T); V staged from the
// pre-transposed vT buffer.  K/V double-buffered on the async engine,
// pipeline unrolled x2 (SEQ/32 = 64 chunks).
// ---------------------------------------------------------------------------
#define ALDK 72    // 144B row
#define ALDV 40    // 80B row
#define ALDP 40

__global__ void __launch_bounds__(32)
flash_attn_wmma(const unsigned short* __restrict__ qkv,   // ROWS x 3D
                const unsigned short* __restrict__ vT,    // [b,h,d,s]
                unsigned short* __restrict__ ctx) {       // ROWS x D
    const int lane  = threadIdx.x;
    const int lmod  = lane & 15, lhalf = lane >> 4;
    const int qt = blockIdx.x;          // 0..127
    const int h  = blockIdx.y;          // 0..11
    const int b  = blockIdx.z;          // 0..1
    const int row0 = b * SEQ + qt * 16;

    __shared__ __attribute__((aligned(16))) unsigned short Ks[2][32 * ALDK];
    __shared__ __attribute__((aligned(16))) unsigned short Vt[2][64 * ALDV];
    __shared__ __attribute__((aligned(16))) unsigned short Ps[16 * ALDP];

    const unsigned short* vhead = vT + ((size_t)b * NHEADS + h) * HEADDIM * SEQ;

    auto stage_kv = [&](int kv, int buf) {
        const unsigned short* kp =
            qkv + (size_t)(b * SEQ + kv + lane) * D3 + DMODEL + h * HEADDIM;
#pragma unroll
        for (int i = 0; i < 8; ++i)
            stage_b128(kp + i * 8, &Ks[buf][lane * ALDK + i * 8]);
#pragma unroll
        for (int rr = 0; rr < 2; ++rr) {
            int d = rr * 32 + lane;
            const unsigned short* vp = vhead + (size_t)d * SEQ + kv;
#pragma unroll
            for (int i = 0; i < 4; ++i)
                stage_b128(vp + i * 8, &Vt[buf][d * ALDV + i * 8]);
        }
    };

    // Q fragments (A operands), 16x64 split into two 16x32 along d.
    FragU qf[2];
    {
        const unsigned short* qp = qkv + (size_t)(row0 + lmod) * D3 + h * HEADDIM;
#pragma unroll
        for (int dk = 0; dk < 2; ++dk) {
            int cb = dk * 32 + lhalf * 8;
            qf[dk].q[0] = *(const v4u*)(qp + cb);
            qf[dk].q[1] = *(const v4u*)(qp + cb + 16);
        }
    }

    v8f cacc[4];
#pragma unroll
    for (int dn = 0; dn < 4; ++dn)
#pragma unroll
        for (int e = 0; e < 8; ++e) cacc[dn][e] = 0.0f;
    float mrun[8], lrun[8];
#pragma unroll
    for (int r = 0; r < 8; ++r) { mrun[r] = -3.0e38f; lrun[r] = 0.0f; }
    const float scale = 0.125f;  // 1/sqrt(64)

    auto attn_step = [&](int buf) {
        // scores: two 16x16 tiles over this kv chunk of 32
        v8f sc[2];
#pragma unroll
        for (int c = 0; c < 2; ++c) {
#pragma unroll
            for (int e = 0; e < 8; ++e) sc[c][e] = 0.0f;
#pragma unroll
            for (int dk = 0; dk < 2; ++dk) {
                FragU bfr;
                int col = c * 16 + lmod;           // kv-local column
                int db  = dk * 32 + lhalf * 8;     // table-order d runs per lane
                bfr.q[0] = *(const v4u*)&Ks[buf][col * ALDK + db];
                bfr.q[1] = *(const v4u*)&Ks[buf][col * ALDK + db + 16];
                sc[c] = wmma_bf16(qf[dk], bfr, sc[c]);
            }
        }
        // online softmax (rows live in 16-lane halves; xor<=8 stays inside)
#pragma unroll
        for (int r = 0; r < 8; ++r) {
            float a = sc[0][r] * scale, c1 = sc[1][r] * scale;
            float m = fmaxf(a, c1);
#pragma unroll
            for (int off = 8; off > 0; off >>= 1)
                m = fmaxf(m, __shfl_xor(m, off, 32));
            float mn = fmaxf(mrun[r], m);
            float e0 = __expf(a - mn), e1 = __expf(c1 - mn);
            sc[0][r] = e0; sc[1][r] = e1;
            float rs = e0 + e1;
#pragma unroll
            for (int off = 8; off > 0; off >>= 1)
                rs += __shfl_xor(rs, off, 32);
            float alpha = __expf(mrun[r] - mn);
            lrun[r] = lrun[r] * alpha + rs;
            mrun[r] = mn;
#pragma unroll
            for (int dn = 0; dn < 4; ++dn) cacc[dn][r] *= alpha;
        }
        // P: C-layout -> LDS row-major -> A fragment
        __syncthreads();
#pragma unroll
        for (int c = 0; c < 2; ++c)
#pragma unroll
            for (int r = 0; r < 8; ++r)
                Ps[(r + lhalf * 8) * ALDP + c * 16 + lmod] = f32_to_bf16(sc[c][r]);
        __syncthreads();
        FragU pf;
        {
            int base = lmod * ALDP + lhalf * 8;
            pf.q[0] = *(const v4u*)&Ps[base];
            pf.q[1] = *(const v4u*)&Ps[base + 16];
        }
        // ctx += P(16x32) * V(32x64)
#pragma unroll
        for (int dn = 0; dn < 4; ++dn) {
            FragU vf;
            int col = dn * 16 + lmod;              // d column
            int kb  = lhalf * 8;                   // kv-local table-order runs
            vf.q[0] = *(const v4u*)&Vt[buf][col * ALDV + kb];
            vf.q[1] = *(const v4u*)&Vt[buf][col * ALDV + kb + 16];
            cacc[dn] = wmma_bf16(pf, vf, cacc[dn]);
        }
    };

    stage_kv(0, 0);                          // prologue
    for (int kv = 0; kv < SEQ; kv += 64) {
        stage_wait();
        __syncthreads();
        stage_kv(kv + 32, 1);                // overlap with WMMA on buf 0
        attn_step(0);

        stage_wait();
        __syncthreads();
        if (kv + 64 < SEQ) stage_kv(kv + 64, 0);
        attn_step(1);
    }

    // ---- normalize + store ctx (bf16) ----
#pragma unroll
    for (int r = 0; r < 8; ++r) {
        float inv = 1.0f / lrun[r];
        int row = row0 + lhalf * 8 + r;
#pragma unroll
        for (int dn = 0; dn < 4; ++dn) {
            int col = h * HEADDIM + dn * 16 + lmod;
            ctx[(size_t)row * DMODEL + col] = f32_to_bf16(cacc[dn][r] * inv);
        }
    }
}

// ---------------------------------------------------------------------------
// Host-side launch
// ---------------------------------------------------------------------------
extern "C" void kernel_launch(void* const* d_in, const int* in_sizes, int n_in,
                              void* d_out, int out_size, void* d_ws, size_t ws_size,
                              hipStream_t stream) {
    (void)in_sizes; (void)n_in; (void)out_size; (void)ws_size;

    const float* x   = (const float*)d_in[0];
    const float* Wq  = (const float*)d_in[1];
    const float* Wk  = (const float*)d_in[2];
    const float* Wv  = (const float*)d_in[3];
    const float* Wo  = (const float*)d_in[4];
    const float* W1  = (const float*)d_in[5];
    const float* W2  = (const float*)d_in[6];
    const float* g1  = (const float*)d_in[7];
    const float* b1  = (const float*)d_in[8];
    const float* g2  = (const float*)d_in[9];
    const float* b2  = (const float*)d_in[10];
    float* out = (float*)d_out;

    char* ws = (char*)d_ws;
    auto carve = [&](size_t bytes) {
        char* p = ws;
        ws += (bytes + 255) & ~(size_t)255;
        return p;
    };
    // all weights stored transposed: Wt[n][k], ld = K
    unsigned short* Wqkv_t  = (unsigned short*)carve((size_t)D3 * DMODEL * 2);
    unsigned short* Wo_t    = (unsigned short*)carve((size_t)DMODEL * DMODEL * 2);
    unsigned short* W1_t    = (unsigned short*)carve((size_t)HIDDEN * DMODEL * 2);
    unsigned short* W2_t    = (unsigned short*)carve((size_t)DMODEL * HIDDEN * 2);
    unsigned short* h1_bf   = (unsigned short*)carve((size_t)ROWS * DMODEL * 2);
    unsigned short* qkv_bf  = (unsigned short*)carve((size_t)ROWS * D3 * 2);
    unsigned short* vT_bf   = (unsigned short*)carve((size_t)ROWS * DMODEL * 2);
    unsigned short* ctx_bf  = (unsigned short*)carve((size_t)ROWS * DMODEL * 2);
    float*          res1    = (float*)carve((size_t)ROWS * DMODEL * 4);
    unsigned short* h2_bf   = (unsigned short*)carve((size_t)ROWS * DMODEL * 2);
    unsigned short* ffn1_bf = (unsigned short*)carve((size_t)ROWS * HIDDEN * 2);

    // 1) weight repack f32 -> bf16, transposed; Wq|Wk|Wv stacked in N.
    const int DD = DMODEL * DMODEL, DH = DMODEL * HIDDEN;
    cvt_t_f32_bf16<<<(DD + 255) / 256, 256, 0, stream>>>(Wq, Wqkv_t,                              DMODEL, DMODEL, DMODEL, DMODEL);
    cvt_t_f32_bf16<<<(DD + 255) / 256, 256, 0, stream>>>(Wk, Wqkv_t + (size_t)DMODEL * DMODEL,    DMODEL, DMODEL, DMODEL, DMODEL);
    cvt_t_f32_bf16<<<(DD + 255) / 256, 256, 0, stream>>>(Wv, Wqkv_t + (size_t)2 * DMODEL * DMODEL,DMODEL, DMODEL, DMODEL, DMODEL);
    cvt_t_f32_bf16<<<(DD + 255) / 256, 256, 0, stream>>>(Wo, Wo_t, DMODEL, DMODEL, DMODEL, DMODEL);
    cvt_t_f32_bf16<<<(DH + 255) / 256, 256, 0, stream>>>(W1, W1_t, DMODEL, HIDDEN, HIDDEN, DMODEL);
    cvt_t_f32_bf16<<<(DH + 255) / 256, 256, 0, stream>>>(W2, W2_t, HIDDEN, DMODEL, DMODEL, HIDDEN);

    // 2) LN1
    layernorm_bf16<<<ROWS, 256, 0, stream>>>(x, g1, b1, h1_bf);

    // 3) fused QKV projection: [4096x768] x [768x2304] -> bf16
    gemm_bf16_wmma<false, false, true><<<dim3(D3 / GBN, ROWS / GBM), 256, 0, stream>>>(
        h1_bf, Wqkv_t, nullptr, nullptr, qkv_bf, ROWS, D3, DMODEL);

    // 3b) V transpose for attention staging
    transpose_v<<<(ROWS * DMODEL + 255) / 256, 256, 0, stream>>>(qkv_bf, vT_bf);

    // 4) flash attention
    flash_attn_wmma<<<dim3(SEQ / 16, NHEADS, BATCH), 32, 0, stream>>>(qkv_bf, vT_bf, ctx_bf);

    // 5) output projection + residual: res1 = x + ctx @ Wo
    gemm_bf16_wmma<false, true, false><<<dim3(DMODEL / GBN, ROWS / GBM), 256, 0, stream>>>(
        ctx_bf, Wo_t, x, res1, nullptr, ROWS, DMODEL, DMODEL);

    // 6) LN2
    layernorm_bf16<<<ROWS, 256, 0, stream>>>(res1, g2, b2, h2_bf);

    // 7) FFN1 + ReLU -> bf16
    gemm_bf16_wmma<true, false, true><<<dim3(HIDDEN / GBN, ROWS / GBM), 256, 0, stream>>>(
        h2_bf, W1_t, nullptr, nullptr, ffn1_bf, ROWS, HIDDEN, DMODEL);

    // 8) FFN2 + residual -> out
    gemm_bf16_wmma<false, true, false><<<dim3(DMODEL / GBN, ROWS / GBM), 256, 0, stream>>>(
        ffn1_bf, W2_t, res1, out, nullptr, ROWS, DMODEL, HIDDEN);
}
